// TorchSDPAttentionLayer_58703613002090
// MI455X (gfx1250) — matrix operand order repaired and checked by reading, not compiled, not run on hardware
//
#include <hip/hip_runtime.h>
#include <hip/hip_bf16.h>

// B=2, H=16, S=2048, D=128 fp32 attention forward producing (Y, weights).
// Strategy (see analysis): bf16 WMMA for both GEMMs, two-pass softmax so the
// 537MB weights output is written exactly once with final normalized values.

#define SEQ 2048
#define DH  128
#define QT  32      // q rows per workgroup
#define KT  64      // kv columns per iteration
#define NBH 32      // B*H

typedef __attribute__((ext_vector_type(16))) __bf16 v16bf;
typedef __attribute__((ext_vector_type(8)))  float  v8f;

// padded LDS row strides (in bf16 elements) to avoid bank conflicts
#define QS_LD 132   // qs[QT][D]
#define KS_LD 68    // ks[D][KT]  (K transposed)
#define VS_LD 132   // vs[KT][D]
#define PS_LD 68    // ps[QT][KT]

__global__ __launch_bounds__(256)
void attn_fwd_kernel(const float* __restrict__ Q,
                     const float* __restrict__ K,
                     const float* __restrict__ V,
                     const int*   __restrict__ mask,
                     float* __restrict__ Yout,
                     float* __restrict__ Wout)
{
    __shared__ __bf16 qs[QT * QS_LD];
    __shared__ __bf16 ks[DH * KS_LD];
    __shared__ __bf16 vs[KT * VS_LD];
    __shared__ __bf16 ps[QT * PS_LD];
    __shared__ float  smax[4][QT];
    __shared__ float  ssum[4][QT];
    __shared__ float  rowM[QT];
    __shared__ float  rowLinv[QT];

    const int tid  = threadIdx.x;
    const int wave = tid >> 5;
    const int lane = tid & 31;

    const int nQT = SEQ / QT;                    // 64 q-tiles per head
    const int bh  = blockIdx.x / nQT;            // 0..31
    const int q0  = (blockIdx.x % nQT) * QT;

    const float* Qp = Q + (size_t)bh * SEQ * DH;
    const float* Kp = K + (size_t)bh * SEQ * DH;
    const float* Vp = V + (size_t)bh * SEQ * DH;
    float* Yp = Yout + (size_t)bh * SEQ * DH;
    float* Wp = Wout + (size_t)bh * SEQ * SEQ;

    const float scale = 0.08838834764831845f;    // 1/sqrt(128)

    // ---- stage Q tile (f32 -> bf16) into LDS ----
    for (int e = tid; e < QT * DH; e += 256) {
        int r = e >> 7, d = e & (DH - 1);
        qs[r * QS_LD + d] = (__bf16)Qp[(size_t)(q0 + r) * DH + d];
    }
    __syncthreads();

    // wave -> (row group, column partition) mapping for the score GEMM
    const int mg    = wave & 1;          // 0: rows 0..15, 1: rows 16..31
    const int ng    = wave >> 1;         // column partition 0..3 (16 cols each)
    const int mrow  = mg * 16;
    const int lhalf = lane >> 4;         // C frag: 0 -> rows m..m+7, 1 -> m+8..m+15
    const int lcol  = lane & 15;         // C frag: column within tile

    // running per-row softmax stats (8 rows/lane, replicated across 16 lanes)
    float rm[8], rl[8];
#pragma unroll
    for (int v = 0; v < 8; ++v) { rm[v] = -3.0e38f; rl[v] = 0.0f; }

    // =================== PASS A: row max / sum(exp) ===================
    for (int kv0 = 0; kv0 < SEQ; kv0 += KT) {
        // stage K tile transposed as bf16: ks[d][kv]
        for (int e = tid; e < KT * DH; e += 256) {
            int d = e & (DH - 1), kv = e >> 7;
            ks[d * KS_LD + kv] = (__bf16)Kp[(size_t)(kv0 + kv) * DH + d];
        }
        __syncthreads();

        // 16x16 score tile at (mrow, kv0 + ng*16), K-dim = 128 in 4 chunks
        v8f c = {};
#pragma unroll
        for (int k0 = 0; k0 < DH; k0 += 32) {
            v16bf a, b;
            const int arow = mrow + lcol;
            const int akb  = k0 + (lhalf << 3);
#pragma unroll
            for (int i = 0; i < 8; ++i) {
                a[i]     = qs[arow * QS_LD + akb + i];
                a[8 + i] = qs[arow * QS_LD + akb + 16 + i];
            }
            const int brow = k0 + (lhalf << 3);      // B frag: lane = N column, K in table order
            const int bn   = ng * 16 + lcol;
#pragma unroll
            for (int i = 0; i < 8; ++i) {
                b[i]     = ks[(brow + i) * KS_LD + bn];
                b[8 + i] = ks[(brow + 16 + i) * KS_LD + bn];
            }
            c = __builtin_amdgcn_wmma_f32_16x16x32_bf16(
                    false, a, false, b, (short)0, c, false, false);
        }

        const int nglob = kv0 + ng * 16 + lcol;
#pragma unroll
        for (int v = 0; v < 8; ++v) {
            const int m = mrow + lhalf * 8 + v;
            float s = c[v] * scale;
            const int mk = mask[(size_t)(q0 + m) * SEQ + nglob];
            s = (mk == 0) ? -1.0e9f : s;
            // tile row-max over the 16 columns (xor masks stay in 16-lane half)
            float tmax = s;
#pragma unroll
            for (int x = 1; x < 16; x <<= 1)
                tmax = fmaxf(tmax, __shfl_xor(tmax, x, 32));
            float ex = __expf(s - tmax);
            float tsum = ex;
#pragma unroll
            for (int x = 1; x < 16; x <<= 1)
                tsum += __shfl_xor(tsum, x, 32);
            // online merge
            const float nm = fmaxf(rm[v], tmax);
            rl[v] = rl[v] * __expf(rm[v] - nm) + tsum * __expf(tmax - nm);
            rm[v] = nm;
        }
        __syncthreads();
    }

    // publish per-wave stats (lane 0 / lane 16 hold replicated row values)
    if (lcol == 0) {
#pragma unroll
        for (int v = 0; v < 8; ++v) {
            const int m = mrow + lhalf * 8 + v;
            smax[ng][m] = rm[v];
            ssum[ng][m] = rl[v];
        }
    }
    __syncthreads();

    // combine the 4 column partitions -> final row max / 1/sum
    if (tid < QT) {
        float M = smax[0][tid];
        M = fmaxf(M, smax[1][tid]);
        M = fmaxf(M, smax[2][tid]);
        M = fmaxf(M, smax[3][tid]);
        float L = 0.0f;
#pragma unroll
        for (int j = 0; j < 4; ++j)
            L += ssum[j][tid] * __expf(smax[j][tid] - M);
        rowM[tid]    = M;
        rowLinv[tid] = 1.0f / L;
    }
    __syncthreads();

    // =================== PASS B: weights + Y ===================
    v8f acc0 = {}, acc1 = {};   // Y tiles (mg, ng) and (mg, ng+4)

    for (int kv0 = 0; kv0 < SEQ; kv0 += KT) {
        // stage K (transposed) and V tiles as bf16
        for (int e = tid; e < KT * DH; e += 256) {
            int d = e & (DH - 1), kv = e >> 7;
            ks[d * KS_LD + kv] = (__bf16)Kp[(size_t)(kv0 + kv) * DH + d];
            vs[kv * VS_LD + d] = (__bf16)Vp[(size_t)(kv0 + kv) * DH + d];
        }
        __syncthreads();

        // recompute the 16x16 score tile
        v8f c = {};
#pragma unroll
        for (int k0 = 0; k0 < DH; k0 += 32) {
            v16bf a, b;
            const int arow = mrow + lcol;
            const int akb  = k0 + (lhalf << 3);
#pragma unroll
            for (int i = 0; i < 8; ++i) {
                a[i]     = qs[arow * QS_LD + akb + i];
                a[8 + i] = qs[arow * QS_LD + akb + 16 + i];
            }
            const int brow = k0 + (lhalf << 3);
            const int bn   = ng * 16 + lcol;
#pragma unroll
            for (int i = 0; i < 8; ++i) {
                b[i]     = ks[(brow + i) * KS_LD + bn];
                b[8 + i] = ks[(brow + 16 + i) * KS_LD + bn];
            }
            c = __builtin_amdgcn_wmma_f32_16x16x32_bf16(
                    false, a, false, b, (short)0, c, false, false);
        }

        // normalize, emit weights (f32) and LDS p-tile (bf16)
        const int nloc  = ng * 16 + lcol;
        const int nglob = kv0 + nloc;
#pragma unroll
        for (int v = 0; v < 8; ++v) {
            const int m = mrow + lhalf * 8 + v;
            float s = c[v] * scale;
            const int mk = mask[(size_t)(q0 + m) * SEQ + nglob];
            s = (mk == 0) ? -1.0e9f : s;
            const float p = __expf(s - rowM[m]) * rowLinv[m];
            Wp[(size_t)(q0 + m) * SEQ + nglob] = p;
            ps[m * PS_LD + nloc] = (__bf16)p;
        }
        __syncthreads();

        // P (32xKT) x V (KTx128): this wave owns Y tiles (mg, ng) and (mg, ng+4)
#pragma unroll
        for (int t = 0; t < 2; ++t) {
            const int n0 = (ng + 4 * t) * 16;
            v8f* accp = t ? &acc1 : &acc0;
#pragma unroll
            for (int k0 = 0; k0 < KT; k0 += 32) {
                v16bf a, b;
                const int arow = mrow + lcol;
                const int akb  = k0 + (lhalf << 3);
#pragma unroll
                for (int i = 0; i < 8; ++i) {
                    a[i]     = ps[arow * PS_LD + akb + i];
                    a[8 + i] = ps[arow * PS_LD + akb + 16 + i];
                }
                const int brow = k0 + (lhalf << 3);
#pragma unroll
                for (int i = 0; i < 8; ++i) {
                    b[i]     = vs[(brow + i) * VS_LD + n0 + lcol];
                    b[8 + i] = vs[(brow + 16 + i) * VS_LD + n0 + lcol];
                }
                *accp = __builtin_amdgcn_wmma_f32_16x16x32_bf16(
                            false, a, false, b, (short)0, *accp, false, false);
            }
        }
        __syncthreads();   // before next iteration overwrites ks/vs/ps
    }

    // ---- store Y ----
#pragma unroll
    for (int t = 0; t < 2; ++t) {
        const v8f acc = t ? acc1 : acc0;
        const int d = (ng + 4 * t) * 16 + lcol;
#pragma unroll
        for (int v = 0; v < 8; ++v) {
            const int m = mrow + lhalf * 8 + v;
            Yp[(size_t)(q0 + m) * DH + d] = acc[v];
        }
    }
}

extern "C" void kernel_launch(void* const* d_in, const int* in_sizes, int n_in,
                              void* d_out, int out_size, void* d_ws, size_t ws_size,
                              hipStream_t stream) {
    const float* Q    = (const float*)d_in[0];
    const float* K    = (const float*)d_in[1];
    const float* V    = (const float*)d_in[2];
    const int*   mask = (const int*)d_in[3];

    float* Y = (float*)d_out;                                  // [B,H,S,D]
    float* W = (float*)d_out + (size_t)NBH * SEQ * DH;         // [B,H,S,S]

    const int blocks = NBH * (SEQ / QT);                       // 2048
    attn_fwd_kernel<<<blocks, 256, 0, stream>>>(Q, K, V, mask, Y, W);
}
